// Classifier_6682969112951
// MI455X (gfx1250) — matrix operand order repaired and checked by reading, not compiled, not run on hardware
//
#include <hip/hip_runtime.h>
#include <stddef.h>

// ---------------------------------------------------------------------------
// MI455X / gfx1250: bf16 WMMA (f32 accumulate) GEMM path, software-pipelined,
// 16x64 tile per wave (spill-free register budget: ~130 VGPRs).
// ---------------------------------------------------------------------------

typedef __attribute__((ext_vector_type(16))) __bf16 bf16x16;
typedef __attribute__((ext_vector_type(8)))  __bf16 bf16x8;
typedef __attribute__((ext_vector_type(8)))  float  f32x8;

union FragU { bf16x16 v; bf16x8 h[2]; };

__device__ __forceinline__ unsigned short f2bf(float f) {
  unsigned u = __float_as_uint(f);
  unsigned r = (u + 0x7FFFu + ((u >> 16) & 1u)) >> 16;  // round-to-nearest-even
  return (unsigned short)r;
}

// A-fragment: two 8-elem chunks at k and k+16 (half-wave K interleave, ISA layout)
__device__ __forceinline__ void ldA(FragU& f, const unsigned short* p, int k) {
  f.h[0] = *(const bf16x8*)(p + k);
  f.h[1] = *(const bf16x8*)(p + k + 16);
}
// B-fragment: two 8-elem chunks at k and k+16 of one weight row (column of B)
__device__ __forceinline__ void ldB(FragU& f, const unsigned short* p, int k) {
  f.h[0] = *(const bf16x8*)(p + k);
  f.h[1] = *(const bf16x8*)(p + k + 16);
}

#define WMMA_BF16(A, B, C) \
  __builtin_amdgcn_wmma_f32_16x16x32_bf16(false, (A).v, false, (B).v, (short)0, (C), false, false)

#define BB 64          // batch
#define SS 300         // seq len
#define SP 320         // seq len padded (20 M-tiles of 16)
#define DM 1024        // d_model
#define PP 128         // d_proj
#define HH 4           // heads
#define NW 768         // fused projection width: [K(128) | V(128) | Q0..Q3(512)]
#define CC 3862        // classes
#define CP 3968        // classes padded to 64

// ---------------- bf16 conversion kernels ----------------------------------

__global__ void k_cvt_x(const float* __restrict__ X, unsigned short* __restrict__ Xbf) {
  size_t i = (size_t)blockIdx.x * 256 + threadIdx.x;      // BB*SP*DM total
  int d = (int)(i % DM);
  int s = (int)((i / DM) % SP);
  int b = (int)(i / ((size_t)DM * SP));
  float v = (s < SS) ? X[((size_t)(b * SS + s)) * DM + d] : 0.0f;
  Xbf[i] = f2bf(v);
}

__global__ void k_cvt_w(const float* __restrict__ Wk, const float* __restrict__ Wv,
                        const float* __restrict__ Wq, unsigned short* __restrict__ Wbf) {
  size_t i = (size_t)blockIdx.x * 256 + threadIdx.x;      // NW*DM total
  int r = (int)(i / DM);
  int d = (int)(i % DM);
  float v;
  if (r < 128)       v = Wk[(size_t)r * DM + d];
  else if (r < 256)  v = Wv[(size_t)(r - 128) * DM + d];
  else               v = Wq[(size_t)(r - 256) * DM + d];  // [H*P, D] flat
  Wbf[i] = f2bf(v);
}

__global__ void k_cvt_wc(const float* __restrict__ Wc, unsigned short* __restrict__ Wcbf) {
  size_t i = (size_t)blockIdx.x * 256 + threadIdx.x;      // CP*PP total
  int c = (int)(i / PP);
  int p = (int)(i % PP);
  float v = (c < CC) ? Wc[(size_t)c * PP + p] : 0.0f;
  Wcbf[i] = f2bf(v);
}

// ---------------- projection GEMM: Y[b] = X[b] @ W^T ------------------------
// One wave per 16x64 tile, double-buffered K loop.
// 64*20*12 = 15360 wave-jobs, 8 waves/block.

__global__ void k_proj_gemm(const unsigned short* __restrict__ Xbf,
                            const unsigned short* __restrict__ Wbf,
                            float* __restrict__ Y) {
  const int wave = threadIdx.x >> 5;
  const int lane = threadIdx.x & 31;
  const int job  = blockIdx.x * 8 + wave;
  if (job >= BB * 20 * 12) return;
  const int b   = job / 240;
  const int rem = job % 240;
  const int mt  = rem / 12;
  const int ng  = rem % 12;
  const int m0 = mt * 16, n0 = ng * 64;
  const int ml = lane & 15, hi = lane >> 4;

  const unsigned short* ap  = Xbf + ((size_t)b * SP + m0 + ml) * DM + hi * 8;
  const unsigned short* bp0 = Wbf + (size_t)(n0 + ml) * DM + hi * 8;
  const unsigned short* bp1 = bp0 + (size_t)16 * DM;
  const unsigned short* bp2 = bp0 + (size_t)32 * DM;
  const unsigned short* bp3 = bp0 + (size_t)48 * DM;

  f32x8 a0 = {}, a1 = {}, a2 = {}, a3 = {};

  FragU xa, wb0, wb1, wb2, wb3;   // "cur" fragment set
  FragU ya, vb0, vb1, vb2, vb3;   // "next" fragment set

  ldA(xa, ap, 0);
  ldB(wb0, bp0, 0); ldB(wb1, bp1, 0); ldB(wb2, bp2, 0); ldB(wb3, bp3, 0);

  for (int k = 0; k < DM; k += 64) {
    // issue loads for k+32 before consuming k
    ldA(ya, ap, k + 32);
    ldB(vb0, bp0, k + 32); ldB(vb1, bp1, k + 32);
    ldB(vb2, bp2, k + 32); ldB(vb3, bp3, k + 32);

    a0 = WMMA_BF16(xa, wb0, a0); a1 = WMMA_BF16(xa, wb1, a1);
    a2 = WMMA_BF16(xa, wb2, a2); a3 = WMMA_BF16(xa, wb3, a3);

    if (k + 64 < DM) {             // issue loads for k+64 before consuming k+32
      ldA(xa, ap, k + 64);
      ldB(wb0, bp0, k + 64); ldB(wb1, bp1, k + 64);
      ldB(wb2, bp2, k + 64); ldB(wb3, bp3, k + 64);
    }

    a0 = WMMA_BF16(ya, vb0, a0); a1 = WMMA_BF16(ya, vb1, a1);
    a2 = WMMA_BF16(ya, vb2, a2); a3 = WMMA_BF16(ya, vb3, a3);
  }

  // C layout: VGPR r <-> row r + 8*hi ; col = lane&15
  float* yr = Y + ((size_t)b * SP + m0 + hi * 8) * NW + n0 + ml;
#pragma unroll
  for (int r = 0; r < 8; ++r) {
    yr[(size_t)r * NW +  0] = a0[r];
    yr[(size_t)r * NW + 16] = a1[r];
    yr[(size_t)r * NW + 32] = a2[r];
    yr[(size_t)r * NW + 48] = a3[r];
  }
}

// ---------------- attention: scores, softmax, weighted sum, relu, bf16 -----

__global__ void k_attn(const float* __restrict__ Y,
                       const float* __restrict__ bk, const float* __restrict__ bv,
                       const float* __restrict__ bq,
                       float* __restrict__ scores_out, float* __restrict__ attnw_out,
                       unsigned short* __restrict__ wsb) {
  const int b = blockIdx.x;
  const int t = threadIdx.x;
  __shared__ float sc[HH * SS];
  __shared__ float red[256];
  const float scale = 0.088388347648318447f;  // 1/sqrt(128)
  const float* Yb = Y + (size_t)b * SP * NW;

  // scores: score[h,s] = dot(q[h,s]+bq, k[s]+bk) * scale
  for (int idx = t; idx < HH * SS; idx += 256) {
    int h = idx / SS, s = idx % SS;
    const float* qp  = Yb + (size_t)s * NW + 256 + h * PP;
    const float* kp  = Yb + (size_t)s * NW;
    const float* bqh = bq + h * PP;
    float acc = 0.f;
    for (int p = 0; p < PP; ++p) acc += (qp[p] + bqh[p]) * (kp[p] + bk[p]);
    float v = acc * scale;
    sc[h * SS + s] = v;
    scores_out[((size_t)b * SS + s) * HH + h] = v;     // [B,S,H]
  }
  __syncthreads();

  // softmax over S for each head
  for (int h = 0; h < HH; ++h) {
    float m = -1e30f;
    for (int s = t; s < SS; s += 256) m = fmaxf(m, sc[h * SS + s]);
    red[t] = m; __syncthreads();
    for (int o = 128; o > 0; o >>= 1) { if (t < o) red[t] = fmaxf(red[t], red[t + o]); __syncthreads(); }
    float mx = red[0]; __syncthreads();
    float sum = 0.f;
    for (int s = t; s < SS; s += 256) { float e = expf(sc[h * SS + s] - mx); sc[h * SS + s] = e; sum += e; }
    red[t] = sum; __syncthreads();
    for (int o = 128; o > 0; o >>= 1) { if (t < o) red[t] += red[t + o]; __syncthreads(); }
    float inv = 1.f / red[0]; __syncthreads();
    for (int s = t; s < SS; s += 256) {
      float p = sc[h * SS + s] * inv;
      sc[h * SS + s] = p;
      attnw_out[((size_t)b * SS + s) * HH + h] = p;    // [B,S,H]
    }
    __syncthreads();
  }

  // ws[h,p] = sum_s p_attn[h,s] * (v[s,p]+bv[p]);  relu; pack bf16 for classifier GEMM
  for (int idx = t; idx < HH * PP; idx += 256) {
    int h = idx >> 7, p = idx & 127;
    const float* vp = Yb + 128 + p;
    const float* pr = sc + h * SS;
    float bvp = bv[p];
    float acc = 0.f;
    for (int s = 0; s < SS; ++s) acc += pr[s] * (vp[(size_t)s * NW] + bvp);
    float r = fmaxf(acc, 0.f);
    wsb[(size_t)(b * HH + h) * PP + p] = f2bf(r);
  }
}

// ---------------- classifier GEMM: [256x128] @ Wc^T[128x3968] ---------------
// 16 * 62 = 992 wave-jobs, 8 waves/block, double-buffered K loop.

__global__ void k_cls_gemm(const unsigned short* __restrict__ A,
                           const unsigned short* __restrict__ Bw,
                           float* __restrict__ Cc) {
  const int wave = threadIdx.x >> 5;
  const int lane = threadIdx.x & 31;
  const int job  = blockIdx.x * 8 + wave;
  if (job >= 16 * 62) return;
  const int mt = job / 62, ng = job % 62;
  const int m0 = mt * 16, n0 = ng * 64;
  const int ml = lane & 15, hi = lane >> 4;

  const unsigned short* ap  = A  + (size_t)(m0 + ml) * PP + hi * 8;
  const unsigned short* bp0 = Bw + (size_t)(n0 + ml) * PP + hi * 8;
  const unsigned short* bp1 = bp0 + (size_t)16 * PP;
  const unsigned short* bp2 = bp0 + (size_t)32 * PP;
  const unsigned short* bp3 = bp0 + (size_t)48 * PP;

  f32x8 a0 = {}, a1 = {}, a2 = {}, a3 = {};
  FragU xa, wb0, wb1, wb2, wb3;
  FragU ya, vb0, vb1, vb2, vb3;

  ldA(xa, ap, 0);
  ldB(wb0, bp0, 0); ldB(wb1, bp1, 0); ldB(wb2, bp2, 0); ldB(wb3, bp3, 0);

  for (int k = 0; k < PP; k += 64) {
    ldA(ya, ap, k + 32);
    ldB(vb0, bp0, k + 32); ldB(vb1, bp1, k + 32);
    ldB(vb2, bp2, k + 32); ldB(vb3, bp3, k + 32);

    a0 = WMMA_BF16(xa, wb0, a0); a1 = WMMA_BF16(xa, wb1, a1);
    a2 = WMMA_BF16(xa, wb2, a2); a3 = WMMA_BF16(xa, wb3, a3);

    if (k + 64 < PP) {
      ldA(xa, ap, k + 64);
      ldB(wb0, bp0, k + 64); ldB(wb1, bp1, k + 64);
      ldB(wb2, bp2, k + 64); ldB(wb3, bp3, k + 64);
    }

    a0 = WMMA_BF16(ya, vb0, a0); a1 = WMMA_BF16(ya, vb1, a1);
    a2 = WMMA_BF16(ya, vb2, a2); a3 = WMMA_BF16(ya, vb3, a3);
  }

  float* cr = Cc + (size_t)(m0 + hi * 8) * CP + n0 + ml;
#pragma unroll
  for (int r = 0; r < 8; ++r) {
    cr[(size_t)r * CP +  0] = a0[r];
    cr[(size_t)r * CP + 16] = a1[r];
    cr[(size_t)r * CP + 32] = a2[r];
    cr[(size_t)r * CP + 48] = a3[r];
  }
}

// ---------------- layernorm + head max/argmax + sigmoid ---------------------

__global__ void k_ln_max(const float* __restrict__ Lg, const float* __restrict__ bc,
                         const float* __restrict__ ln_a, const float* __restrict__ ln_b,
                         float* __restrict__ vid_probs, float* __restrict__ attn_idc) {
  const int b = blockIdx.x;
  const int t = threadIdx.x;
  __shared__ float red[256];
  __shared__ float meanS[HH], invS[HH];

  for (int h = 0; h < HH; ++h) {
    const float* L = Lg + (size_t)(b * HH + h) * CP;
    float s = 0.f, sq = 0.f;
    for (int c = t; c < CC; c += 256) { float x = L[c] + bc[c]; s += x; sq += x * x; }
    red[t] = s; __syncthreads();
    for (int o = 128; o > 0; o >>= 1) { if (t < o) red[t] += red[t + o]; __syncthreads(); }
    float mean = red[0] / (float)CC; __syncthreads();
    red[t] = sq; __syncthreads();
    for (int o = 128; o > 0; o >>= 1) { if (t < o) red[t] += red[t + o]; __syncthreads(); }
    float sumsq = red[0]; __syncthreads();
    float var = (sumsq - (float)CC * mean * mean) / (float)(CC - 1);
    float stdv = sqrtf(fmaxf(var, 0.f));
    if (t == 0) { meanS[h] = mean; invS[h] = 1.f / (stdv + 1e-6f); }
    __syncthreads();
  }

  for (int c = t; c < CC; c += 256) {
    float a = ln_a[c], bb2 = ln_b[c], bcc = bc[c];
    float best = -1e30f; int bi = 0;
#pragma unroll
    for (int h = 0; h < HH; ++h) {
      float x = Lg[(size_t)(b * HH + h) * CP + c] + bcc;
      float y = a * (x - meanS[h]) * invS[h] + bb2;
      if (y > best) { best = y; bi = h; }            // first-occurrence argmax
    }
    vid_probs[(size_t)b * CC + c] = 1.f / (1.f + expf(-best));
    attn_idc[(size_t)b * CC + c] = (float)bi;
  }
}

// ---------------- conv regularizer scalar -----------------------------------

__global__ void k_convloss(const float* __restrict__ Wc, const float* __restrict__ bc,
                           float* __restrict__ out) {
  const int t = threadIdx.x;
  __shared__ float cp[CC];
  __shared__ float red[256];
  for (int c = t; c < CC; c += 256) {
    const float* w = Wc + (size_t)c * PP;
    float s = bc[c];
    for (int p = 0; p < PP; ++p) s += w[p];
    cp[c] = s;
  }
  __syncthreads();
  float m = -1e30f;
  for (int c = t; c < CC; c += 256) m = fmaxf(m, cp[c]);
  red[t] = m; __syncthreads();
  for (int o = 128; o > 0; o >>= 1) { if (t < o) red[t] = fmaxf(red[t], red[t + o]); __syncthreads(); }
  float mx = red[0]; __syncthreads();
  float s = 0.f;
  for (int c = t; c < CC; c += 256) { float e = expf(cp[c] - mx); cp[c] = e; s += e; }
  red[t] = s; __syncthreads();
  for (int o = 128; o > 0; o >>= 1) { if (t < o) red[t] += red[t + o]; __syncthreads(); }
  float inv = 1.f / red[0]; __syncthreads();
  float sp = 0.f, spq = 0.f;
  for (int c = t; c < CC; c += 256) { float p = cp[c] * inv; sp += p; spq += p * p; }
  red[t] = sp; __syncthreads();
  for (int o = 128; o > 0; o >>= 1) { if (t < o) red[t] += red[t + o]; __syncthreads(); }
  float tot = red[0]; __syncthreads();
  red[t] = spq; __syncthreads();
  for (int o = 128; o > 0; o >>= 1) { if (t < o) red[t] += red[t + o]; __syncthreads(); }
  float totq = red[0];
  if (t == 0) {
    float mean = tot / (float)CC;
    float var = (totq - (float)CC * mean * mean) / (float)(CC - 1);
    float stdv = sqrtf(fmaxf(var, 0.f));
    stdv = fminf(fmaxf(stdv, 1e-9f), 1e9f);
    out[0] = (float)BB * stdv;
  }
}

// ---------------- launch -----------------------------------------------------

extern "C" void kernel_launch(void* const* d_in, const int* in_sizes, int n_in,
                              void* d_out, int out_size, void* d_ws, size_t ws_size,
                              hipStream_t stream) {
  const float* X    = (const float*)d_in[0];
  const float* Wq   = (const float*)d_in[1];
  const float* bq   = (const float*)d_in[2];
  const float* Wk   = (const float*)d_in[3];
  const float* bk   = (const float*)d_in[4];
  const float* Wv   = (const float*)d_in[5];
  const float* bv   = (const float*)d_in[6];
  const float* Wc   = (const float*)d_in[7];
  const float* bc   = (const float*)d_in[8];
  const float* ln_a = (const float*)d_in[9];
  const float* ln_b = (const float*)d_in[10];

  // workspace layout (all offsets 256B-aligned), ~111.6 MB total
  char* ws = (char*)d_ws;
  unsigned short* Xbf  = (unsigned short*)(ws + 0);           //  64*320*1024 bf16
  unsigned short* Wbf  = (unsigned short*)(ws + 41943040);    //  768*1024 bf16
  unsigned short* Wcbf = (unsigned short*)(ws + 43515904);    //  3968*128 bf16
  float*          Y    = (float*)(ws + 44531712);             //  64*320*768 f32
  unsigned short* wsb  = (unsigned short*)(ws + 107446272);   //  256*128 bf16
  float*          Lg   = (float*)(ws + 107511808);            //  256*3968 f32

  float* out       = (float*)d_out;
  float* vid_probs = out;             // [64,3862]
  float* attn_idc  = out + 247168;    // [64,3862] (float-cast index)
  float* scores    = out + 494336;    // [64,300,4]
  float* attnw     = out + 571136;    // [64,300,4]
  float* closs     = out + 647936;    // scalar

  k_cvt_x    <<<81920, 256, 0, stream>>>(X, Xbf);
  k_cvt_w    <<<3072,  256, 0, stream>>>(Wk, Wv, Wq, Wbf);
  k_cvt_wc   <<<1984,  256, 0, stream>>>(Wc, Wcbf);
  k_proj_gemm<<<1920,  256, 0, stream>>>(Xbf, Wbf, Y);
  k_attn     <<<64,    256, 0, stream>>>(Y, bk, bv, bq, scores, attnw, wsb);
  k_cls_gemm <<<124,   256, 0, stream>>>(wsb, Wcbf, Lg);
  k_ln_max   <<<64,    256, 0, stream>>>(Lg, bc, ln_a, ln_b, vid_probs, attn_idc);
  k_convloss <<<1,     256, 0, stream>>>(Wc, bc, closs);
}
